// PhraseConsensusHead_54417235640751
// MI455X (gfx1250) — matrix operand order repaired and checked by reading, not compiled, not run on hardware
//
#include <hip/hip_runtime.h>
#include <hip/hip_bf16.h>
#include <math.h>

// ---------------------------------------------------------------------------
// CDNA5 (gfx1250, wave32) WMMA types: v_wmma_f32_16x16x32_bf16
// ---------------------------------------------------------------------------
typedef __bf16 bf16_t;
typedef __attribute__((ext_vector_type(16))) __bf16 v16bf;
typedef __attribute__((ext_vector_type(8)))  float  v8f;

__device__ __forceinline__ unsigned bf16_rne_bits(float f) {
    unsigned u = __builtin_bit_cast(unsigned, f);
    u += 0x7FFFu + ((u >> 16) & 1u);
    return u >> 16;                       // low 16 bits = bf16
}
__device__ __forceinline__ float hi_as_f32(unsigned hibits16) {
    return __builtin_bit_cast(float, hibits16 << 16);
}

// ---------------------------------------------------------------------------
// One-time fp32 -> (bf16 hi, bf16 lo) decomposition into PRE-SWIZZLED
// WMMA-fragment layout:  dst[ tile(=mt*KT+kb) ][ lane(0..31) ][ e(0..15) ]
//   PAT=0 (A operand, 16x32 tile): lane row = mt*16 + (lane&15),
//        k = kb*32 + 8*(lane>>4) + {0..7} U {16..23}
//   PAT=1 (B operand from W[N,K], 32x16 tile): lane row = nt*16 + (lane&15),
//        k = kb*32 + 8*(lane>>4) + {0..7} U {16..23}
// GEMM then loads each lane's whole fragment as one aligned 32-byte vector.
// ---------------------------------------------------------------------------
template <int PAT>
__global__ __launch_bounds__(256)
void pch_decompose(const float* __restrict__ src,
                   unsigned* __restrict__ dhi, unsigned* __restrict__ dlo,
                   int R, int K)
{
    const int KT = K >> 5;
    const int ntile = (R >> 4) * KT;
    const int tid  = blockIdx.x * 256 + threadIdx.x;
    const int lane = tid & 31;
    const int tile = tid >> 5;
    if (tile >= ntile) return;

    const int kb  = tile % KT;
    const int mt  = tile / KT;
    const int row = mt * 16 + (lane & 15);
    const float* p = src + (size_t)row * (size_t)K + (size_t)kb * 32
                   + 8 * (lane >> 4);

    float f[16];
    if (PAT) {
        const float4* q = (const float4*)p;
        float4 a = q[0], b = q[1], c = q[4], d = q[5];
        f[0]=a.x; f[1]=a.y; f[2]=a.z; f[3]=a.w;
        f[4]=b.x; f[5]=b.y; f[6]=b.z; f[7]=b.w;
        f[8]=c.x; f[9]=c.y; f[10]=c.z; f[11]=c.w;
        f[12]=d.x; f[13]=d.y; f[14]=d.z; f[15]=d.w;
    } else {
        const float4* q = (const float4*)p;
        float4 a = q[0], b = q[1], c = q[4], d = q[5];   // {0..7} and {16..23}
        f[0]=a.x; f[1]=a.y; f[2]=a.z; f[3]=a.w;
        f[4]=b.x; f[5]=b.y; f[6]=b.z; f[7]=b.w;
        f[8]=c.x; f[9]=c.y; f[10]=c.z; f[11]=c.w;
        f[12]=d.x; f[13]=d.y; f[14]=d.z; f[15]=d.w;
    }

    unsigned hp[8], lp[8];
    #pragma unroll
    for (int e = 0; e < 8; ++e) {
        float a = f[2 * e], b = f[2 * e + 1];
        unsigned ha = bf16_rne_bits(a), hb = bf16_rne_bits(b);
        float la = a - hi_as_f32(ha), lb = b - hi_as_f32(hb);
        hp[e] = ha | (hb << 16);
        lp[e] = bf16_rne_bits(la) | (bf16_rne_bits(lb) << 16);
    }

    // 16 bf16 = 8 uints = 32 bytes per lane; lane-contiguous -> coalesced
    const size_t off = (size_t)tile * 256 + (size_t)lane * 8;  // uint units
    uint4* oh = (uint4*)(dhi + off);
    uint4* ol = (uint4*)(dlo + off);
    oh[0] = make_uint4(hp[0], hp[1], hp[2], hp[3]);
    oh[1] = make_uint4(hp[4], hp[5], hp[6], hp[7]);
    ol[0] = make_uint4(lp[0], lp[1], lp[2], lp[3]);
    ol[1] = make_uint4(lp[4], lp[5], lp[6], lp[7]);
}

// ---------------------------------------------------------------------------
// GEMM: C[M,N] = A[M,K] @ W[N,K]^T + bias[N], operands pre-split bf16 hi/lo.
// Error-compensated: hi*hi + lo*hi + hi*lo (3 WMMAs / 32-K block) ~ fp32.
// Wave tile 16(M) x 64(N): 4 accumulators, A fragment reused 4x.
// Block 256 = 8 waves -> 128(M) x 64(N). Inner loop is loads + WMMA only.
// ---------------------------------------------------------------------------
__global__ __launch_bounds__(256)
void pch_gemm_sw(const bf16_t* __restrict__ Ahi, const bf16_t* __restrict__ Alo,
                 const bf16_t* __restrict__ Whi, const bf16_t* __restrict__ Wlo,
                 const float* __restrict__ bias, float* __restrict__ C,
                 int M, int N, int K)
{
    const int lane = threadIdx.x & 31;
    const int wave = threadIdx.x >> 5;
    const int half = lane >> 4;
    const int l16  = lane & 15;
    const int KT   = K >> 5;

    const int m0 = blockIdx.y * 128 + wave * 16;
    const int n0 = blockIdx.x * 64;
    if (m0 >= M) return;                   // wave-uniform guard

    v8f acc[4];
    acc[0] = (v8f)(0.0f); acc[1] = (v8f)(0.0f);
    acc[2] = (v8f)(0.0f); acc[3] = (v8f)(0.0f);

    const size_t lane16  = (size_t)lane * 16;
    const size_t jstride = (size_t)KT * 512;
    const bf16_t* pah = Ahi + (size_t)(m0 >> 4) * jstride + lane16;
    const bf16_t* pal = Alo + (size_t)(m0 >> 4) * jstride + lane16;
    const bf16_t* pwh = Whi + (size_t)(n0 >> 4) * jstride + lane16;
    const bf16_t* pwl = Wlo + (size_t)(n0 >> 4) * jstride + lane16;

    #pragma unroll 2
    for (int kb = 0; kb < KT; ++kb) {
        const size_t ko = (size_t)kb * 512;
        __builtin_prefetch(pah + ko + 4 * 512, 0, 1);   // global_prefetch_b8
        v16bf a_hi = *(const v16bf*)(pah + ko);
        v16bf a_lo = *(const v16bf*)(pal + ko);
        #pragma unroll
        for (int j = 0; j < 4; ++j) {
            v16bf b_hi = *(const v16bf*)(pwh + j * jstride + ko);
            v16bf b_lo = *(const v16bf*)(pwl + j * jstride + ko);
            acc[j] = __builtin_amdgcn_wmma_f32_16x16x32_bf16(
                false, a_hi, false, b_hi, (short)0, acc[j], false, false);
            acc[j] = __builtin_amdgcn_wmma_f32_16x16x32_bf16(
                false, a_lo, false, b_hi, (short)0, acc[j], false, false);
            acc[j] = __builtin_amdgcn_wmma_f32_16x16x32_bf16(
                false, a_hi, false, b_lo, (short)0, acc[j], false, false);
        }
    }

    // C/D layout: VGPR v, lane -> row = m0 + v + 8*half, col = n0 + 16j + l16
    #pragma unroll
    for (int j = 0; j < 4; ++j) {
        const int col = n0 + 16 * j + l16;
        const float bv = bias[col];
        #pragma unroll
        for (int v = 0; v < 8; ++v) {
            const int row = m0 + v + 8 * half;
            C[(size_t)row * (size_t)N + col] = acc[j][v] + bv;  // scalar b32
        }
    }
}

// ---------------------------------------------------------------------------
// chunk mean: cm[bn, d] = mean_{c<16} tp[bn*16 + c, d]
// ---------------------------------------------------------------------------
__global__ __launch_bounds__(256)
void pch_chunk_mean(const float* __restrict__ tp, float* __restrict__ cm,
                    int total, int D)
{
    int i = blockIdx.x * blockDim.x + threadIdx.x;
    if (i >= total) return;
    int d  = i & (D - 1);
    int bn = i / D;
    const float* p = tp + (size_t)bn * 16u * (size_t)D + d;
    float s = 0.0f;
    #pragma unroll
    for (int c = 0; c < 16; ++c) s += p[(size_t)c * D];
    cm[i] = s * (1.0f / 16.0f);
}

__global__ void pch_zero_score(float* score)
{
    if (threadIdx.x == 0 && blockIdx.x == 0) *score = 0.0f;
}

// ---------------------------------------------------------------------------
// Per token (one wave32 / token): gate, feedback, cosine-similarity mean.
// ---------------------------------------------------------------------------
__global__ __launch_bounds__(256)
void pch_gate_feedback_score(const float* __restrict__ x,
                             const float* __restrict__ Wg,
                             const float* __restrict__ bg,
                             const float* __restrict__ tp,
                             const float* __restrict__ pc,
                             float* __restrict__ fb,
                             float* __restrict__ score,
                             int D, int Stot)
{
    const int t    = blockIdx.x * 8 + (threadIdx.x >> 5);
    const int lane = threadIdx.x & 31;
    if (t >= Stot) return;
    const int r = t >> 4;

    const float* xr = x  + (size_t)t * D;
    const float* tr = tp + (size_t)t * D;
    const float* cr = pc + (size_t)r * D;

    float sg = 0.f, sn = 0.f, st = 0.f, sc = 0.f;
    for (int i = lane; i < D; i += 32) {
        float xv = xr[i], tv = tr[i], cv = cr[i], wv = Wg[i];
        sg += xv * wv;
        sn += tv * cv;
        st += tv * tv;
        sc += cv * cv;
    }
    #pragma unroll
    for (int off = 16; off; off >>= 1) {
        sg += __shfl_xor(sg, off, 32);
        sn += __shfl_xor(sn, off, 32);
        st += __shfl_xor(st, off, 32);
        sc += __shfl_xor(sc, off, 32);
    }

    const float g = 1.0f / (1.0f + __expf(-(sg + bg[0])));
    float* fr = fb + (size_t)t * D;
    for (int i = lane; i < D; i += 32) fr[i] = g * cr[i];

    if (lane == 0) {
        float den = fmaxf(sqrtf(st), 1e-8f) * fmaxf(sqrtf(sc), 1e-8f);
        atomicAdd(score, (sn / den) * (1.0f / (float)Stot));
    }
}

// ---------------------------------------------------------------------------
// launch: all scratch lives in not-yet-written regions of d_out.
// ---------------------------------------------------------------------------
extern "C" void kernel_launch(void* const* d_in, const int* in_sizes, int n_in,
                              void* d_out, int out_size, void* d_ws, size_t ws_size,
                              hipStream_t stream)
{
    (void)n_in; (void)d_ws; (void)ws_size; (void)out_size;

    const float* x  = (const float*)d_in[0];
    const float* Wp = (const float*)d_in[1];
    const float* bp = (const float*)d_in[2];
    const float* Wc = (const float*)d_in[3];
    const float* bc = (const float*)d_in[4];
    const float* Wg = (const float*)d_in[5];
    const float* bg = (const float*)d_in[6];

    const int D  = in_sizes[2];          // 1024
    const int BS = in_sizes[0] / D;      // 16384 tokens
    const int BN = BS / 16;              // 1024 chunks

    float* out = (float*)d_out;
    const size_t PC_N = (size_t)BN * D;  // 1,048,576 floats (4 MB)
    const size_t FB_N = (size_t)BS * D;  // 16,777,216 floats (64 MB)
    float* pcp   = out;                  // phrase_consensus
    float* fbp   = out + PC_N;           // feedback (scratch until last kernel)
    float* score = out + PC_N + FB_N;
    float* tpp   = score + 1;            // token_proposals (4B-aligned)

    // --- scratch inside d_out -------------------------------------------------
    // Phase A (until GEMM1 done): x hi/lo fill fb region; Wp hi/lo fill pc region
    bf16_t* x_hi  = (bf16_t*)fbp;                       // 32 MB
    bf16_t* x_lo  = (bf16_t*)(fbp + FB_N / 2);          // 32 MB
    bf16_t* wp_hi = (bf16_t*)pcp;                       // 2 MB
    bf16_t* wp_lo = (bf16_t*)(pcp + PC_N / 2);          // 2 MB
    // Phase B (after GEMM1): fb region reused for cm + decompositions
    float*  cm    = fbp;                                // 4 MB
    bf16_t* cm_hi = (bf16_t*)(fbp + PC_N);              // +4 MB
    bf16_t* cm_lo = (bf16_t*)(fbp + PC_N + PC_N / 2);   // +6 MB
    bf16_t* wc_hi = (bf16_t*)(fbp + 2 * PC_N);          // +8 MB
    bf16_t* wc_lo = (bf16_t*)(fbp + 2 * PC_N + PC_N / 2); // +10 MB

    const int KT = D / 32;

    // 1) decompose x (A-pattern) and Wp (B-pattern)
    {
        int thr = (BS / 16) * KT * 32;
        pch_decompose<0><<<thr / 256, 256, 0, stream>>>(x, (unsigned*)x_hi,
                                                        (unsigned*)x_lo, BS, D);
        int thw = (D / 16) * KT * 32;
        pch_decompose<1><<<thw / 256, 256, 0, stream>>>(Wp, (unsigned*)wp_hi,
                                                        (unsigned*)wp_lo, D, D);
    }
    // 2) token_proposals = x @ Wp^T + bp   [16384 x 1024]
    {
        dim3 grid(D / 64, BS / 128);
        pch_gemm_sw<<<grid, 256, 0, stream>>>(x_hi, x_lo, wp_hi, wp_lo, bp, tpp,
                                              BS, D, D);
    }
    // 3) chunk mean -> cm (overwrites x_hi stash; no longer needed)
    pch_chunk_mean<<<((int)PC_N + 255) / 256, 256, 0, stream>>>(tpp, cm,
                                                                (int)PC_N, D);
    // 4) decompose cm (A-pattern) and Wc (B-pattern)
    {
        int thr = (BN / 16) * KT * 32;
        pch_decompose<0><<<thr / 256, 256, 0, stream>>>(cm, (unsigned*)cm_hi,
                                                        (unsigned*)cm_lo, BN, D);
        int thw = (D / 16) * KT * 32;
        pch_decompose<1><<<thw / 256, 256, 0, stream>>>(Wc, (unsigned*)wc_hi,
                                                        (unsigned*)wc_lo, D, D);
    }
    // 5) phrase_consensus = cm @ Wc^T + bc   [1024 x 1024] (overwrites wp stash)
    {
        dim3 grid(D / 64, BN / 128);
        pch_gemm_sw<<<grid, 256, 0, stream>>>(cm_hi, cm_lo, wc_hi, wc_lo, bc,
                                              pcp, BN, D, D);
    }
    // 6) score = 0, then gate + feedback + cosine mean (overwrites fb scratch)
    pch_zero_score<<<1, 64, 0, stream>>>(score);
    pch_gate_feedback_score<<<BS / 8, 256, 0, stream>>>(x, Wg, bg, tpp, pcp,
                                                        fbp, score, D, BS);
}
